// MultiQueryAttention_38474317038262
// MI455X (gfx1250) — matrix operand order repaired and checked by reading, compiled, not run on hardware
//
#include <hip/hip_runtime.h>

typedef __attribute__((ext_vector_type(16))) __bf16 bf16x16;
typedef __attribute__((ext_vector_type(8)))  __bf16 bf16x8;
typedef __attribute__((ext_vector_type(8)))  float  f32x8;
typedef __attribute__((ext_vector_type(4)))  float  f32x4;
typedef __attribute__((ext_vector_type(4)))  int    v4i;

typedef __attribute__((address_space(1))) v4i* glob_v4i;
typedef __attribute__((address_space(3))) v4i* lds_v4i;

union BF16Frag { bf16x16 v; bf16x8 h[2]; };

// async 16B global->LDS copy (GLOBAL_LOAD_ASYNC_TO_LDS_B128, ASYNCcnt-tracked)
#define ASYNC_CP16(gsrc, ldst)                                    \
  __builtin_amdgcn_global_load_async_to_lds_b128(                 \
      (glob_v4i)(size_t)(gsrc), (lds_v4i)(ldst), 0, 0)

// 16-lane (half-wave) max reduction on the VALU via v_permlane16_b32
// (xor-butterfly selector nibbles), keeping the DS pipe free for wmma feeds.
__device__ inline float rowmax16(float x) {
  union { float f; unsigned u; } a, b;
  a.f = x;
  b.u = __builtin_amdgcn_permlane16(a.u, a.u, 0x67452301u, 0xEFCDAB89u, false, false); // xor 1
  a.f = fmaxf(a.f, b.f);
  b.u = __builtin_amdgcn_permlane16(a.u, a.u, 0x54761032u, 0xDCFE98BAu, false, false); // xor 2
  a.f = fmaxf(a.f, b.f);
  b.u = __builtin_amdgcn_permlane16(a.u, a.u, 0x32107654u, 0xBA98FEDCu, false, false); // xor 4
  a.f = fmaxf(a.f, b.f);
  b.u = __builtin_amdgcn_permlane16(a.u, a.u, 0xFEDCBA98u, 0x76543210u, false, false); // xor 8
  return fmaxf(a.f, b.f);
}

// ---------------------------------------------------------------------------
// C[M,N] = (A[M,K] @ W[K,N] + bias[N]) * oscale
// InT = float (convert to bf16 on the fly) or __bf16 (async copy to LDS).
// Block: 256 thr = 8 waves (4x2), block tile 256(M) x 128(N), K step 32.
// Each wave owns a 64x64 tile = 4x4 wmma tiles.
// ---------------------------------------------------------------------------
template <typename InT, typename OutT>
__global__ __launch_bounds__(256)
void gemm_bf16_wmma(const InT* __restrict__ A, const float* __restrict__ W,
                    const float* __restrict__ bias, OutT* __restrict__ C,
                    int M, int N, int K, float oscale) {
  __shared__ __bf16 As[256][40];   // [m][k], rows padded 32->40 (80B)
  __shared__ __bf16 Bts[128][40];  // [n][k] (W tile transposed)

  const int tid  = threadIdx.x;
  const int lane = tid & 31;
  const int wid  = tid >> 5;
  const int waveM = wid & 3;       // 4 waves over M (64 rows each)
  const int waveN = wid >> 2;      // 2 waves over N (64 cols each)
  const int half = lane >> 4;
  const int l16  = lane & 15;

  const int mBase = blockIdx.y * 256;
  const int nBase = blockIdx.x * 128;

  f32x8 acc[4][4] = {};

  for (int k0 = 0; k0 < K; k0 += 32) {
    // ---- A tile: 256x32 ----
    if constexpr (sizeof(InT) == 4) {          // fp32 -> bf16, 2048 quads/256thr
#pragma unroll
      for (int i = 0; i < 8; ++i) {
        int q  = tid + i * 256;
        int r  = q >> 3;
        int qc = (q & 7) << 2;
        f32x4 vv = *(const f32x4*)((const float*)A + (size_t)(mBase + r) * K + k0 + qc);
        As[r][qc + 0] = (__bf16)vv.x; As[r][qc + 1] = (__bf16)vv.y;
        As[r][qc + 2] = (__bf16)vv.z; As[r][qc + 3] = (__bf16)vv.w;
      }
    } else {                                   // bf16: async global->LDS copy
#pragma unroll
      for (int i = 0; i < 4; ++i) {
        int q  = tid + i * 256;
        int r  = q >> 2;
        int oc = (q & 3) << 3;
        ASYNC_CP16((const __bf16*)A + (size_t)(mBase + r) * K + k0 + oc,
                   &As[r][oc]);
      }
    }
    // ---- W tile 32x128 fp32, transposed into Bts[n][k]: 1024 quads ----
#pragma unroll
    for (int i = 0; i < 4; ++i) {
      int q  = tid + i * 256;
      int r  = q >> 5;            // k row 0..31 (32 quads per row)
      int nc = (q & 31) << 2;     // n col 0..124
      f32x4 vv = *(const f32x4*)(W + (size_t)(k0 + r) * N + nBase + nc);
      Bts[nc + 0][r] = (__bf16)vv.x; Bts[nc + 1][r] = (__bf16)vv.y;
      Bts[nc + 2][r] = (__bf16)vv.z; Bts[nc + 3][r] = (__bf16)vv.w;
    }
    if constexpr (sizeof(InT) == 2) __builtin_amdgcn_s_wait_asynccnt(0);
    __syncthreads();

    BF16Frag af[4], bf[4];
#pragma unroll
    for (int mt = 0; mt < 4; ++mt) {
      const __bf16* src = &As[waveM * 64 + mt * 16 + l16][0];
      af[mt].h[0] = *(const bf16x8*)(src + 16 * half);      // K = 16h+0..7
      af[mt].h[1] = *(const bf16x8*)(src + 16 * half + 8);  // K = 16h+8..15
    }
#pragma unroll
    for (int nt = 0; nt < 4; ++nt) {
      const __bf16* src = &Bts[waveN * 64 + nt * 16 + l16][16 * half];
      bf[nt].h[0] = *(const bf16x8*)(src);                  // K = 16h+0..7
      bf[nt].h[1] = *(const bf16x8*)(src + 8);              // K = 16h+8..15
    }
#pragma unroll
    for (int mt = 0; mt < 4; ++mt)
#pragma unroll
      for (int nt = 0; nt < 4; ++nt)
        acc[mt][nt] = __builtin_amdgcn_wmma_f32_16x16x32_bf16(
            false, af[mt].v, false, bf[nt].v, (short)0, acc[mt][nt], false, false);
    __syncthreads();
  }

#pragma unroll
  for (int nt = 0; nt < 4; ++nt) {
    int colg = nBase + waveN * 64 + nt * 16 + l16;
    float bv = bias[colg];
#pragma unroll
    for (int mt = 0; mt < 4; ++mt) {
      int rowg0 = mBase + waveM * 64 + mt * 16 + 8 * half;  // C row = v + 8*half
#pragma unroll
      for (int v = 0; v < 8; ++v) {
        float val = (acc[mt][nt][v] + bv) * oscale;
        C[(size_t)(rowg0 + v) * N + colg] = (OutT)val;
      }
    }
  }
}

// ---------------------------------------------------------------------------
// Flash-style MQA. Block = 512 thr = 16 waves; wave == head; all 16 heads share
// one 64-key K/V LDS chunk (MQA reuse 16x). Row max via v_permlane16 butterfly,
// row sums via an extra wmma against an all-ones B fragment. K chunk staged
// with async global->LDS loads; V transposed through VGPRs.
// K pre-scaled by 1/sqrt(d) in its projection GEMM.
// ---------------------------------------------------------------------------
__global__ __launch_bounds__(512)
void mqa_flash(const __bf16* __restrict__ Q,   // [B*S, 2048]
               const __bf16* __restrict__ Kv,  // [B*S, 128] (pre-scaled)
               const __bf16* __restrict__ Vv,  // [B*S, 128]
               __bf16* __restrict__ O,         // [B*S, 2048]
               int S) {
  __shared__ __bf16 Ks[64][136];      // [key][d]
  __shared__ __bf16 Vts[128][72];     // [d][key]
  __shared__ __bf16 Ps[16][16][40];   // per-wave P tile, reused 2x per chunk

  const int tid  = threadIdx.x;
  const int lane = tid & 31;
  const int h    = tid >> 5;          // head
  const int half = lane >> 4;
  const int l16  = lane & 15;

  const int row0   = blockIdx.x * 16;     // flattened [B*S] query row base
  const int kvBase = (row0 / S) * S;

  BF16Frag qf[4];
  {
    const __bf16* qrow = Q + (size_t)(row0 + l16) * 2048 + h * 128;
#pragma unroll
    for (int s = 0; s < 4; ++s) {
      qf[s].h[0] = *(const bf16x8*)(qrow + s * 32 + 16 * half);
      qf[s].h[1] = *(const bf16x8*)(qrow + s * 32 + 16 * half + 8);
    }
  }

  BF16Frag ones;
#pragma unroll
  for (int i = 0; i < 16; ++i) ones.v[i] = (__bf16)1.0f;

  f32x8 acc[8] = {};
  f32x8 accl = {};                    // row-sum accumulator (cols replicated)
  float run_m[8];
#pragma unroll
  for (int v = 0; v < 8; ++v) run_m[v] = -1e30f;

  for (int k0 = 0; k0 < S; k0 += 64) {
    // K chunk: async copy straight into LDS (no VGPR staging, no ds_store).
    // V chunk: load to VGPRs and transpose-scatter into LDS.
#pragma unroll
    for (int i = 0; i < 2; ++i) {
      int idx = (tid + i * 512) * 8;
      int r = idx >> 7, c = idx & 127;
      ASYNC_CP16(Kv + (size_t)(kvBase + k0 + r) * 128 + c, &Ks[r][c]);
      bf16x8 vv = *(const bf16x8*)(Vv + (size_t)(kvBase + k0 + r) * 128 + c);
#pragma unroll
      for (int j = 0; j < 8; ++j) Vts[c + j][r] = vv[j];
    }
    __builtin_amdgcn_s_wait_asynccnt(0);
    __syncthreads();

    // scores: 16(q) x 64(key)
    f32x8 cs[4] = {};
#pragma unroll
    for (int nt = 0; nt < 4; ++nt)
#pragma unroll
      for (int s = 0; s < 4; ++s) {
        BF16Frag kb;
        const __bf16* src = &Ks[nt * 16 + l16][s * 32 + 16 * half];
        kb.h[0] = *(const bf16x8*)(src);
        kb.h[1] = *(const bf16x8*)(src + 8);
        cs[nt] = __builtin_amdgcn_wmma_f32_16x16x32_bf16(
            false, qf[s].v, false, kb.v, (short)0, cs[nt], false, false);
      }

    // online softmax: row max (permlane16 butterfly) + rescale
    float pA[8], pB[8];
#pragma unroll
    for (int v = 0; v < 8; ++v) {
      float mloc = rowmax16(fmaxf(fmaxf(cs[0][v], cs[1][v]),
                                  fmaxf(cs[2][v], cs[3][v])));
      float mnew = fmaxf(run_m[v], mloc);
      float alpha = __expf(run_m[v] - mnew);
      run_m[v] = mnew;
      accl[v] *= alpha;
#pragma unroll
      for (int dt = 0; dt < 8; ++dt) acc[dt][v] *= alpha;
      float e0 = __expf(cs[0][v] - mnew);
      float e1 = __expf(cs[1][v] - mnew);
      pA[v] = __expf(cs[2][v] - mnew);
      pB[v] = __expf(cs[3][v] - mnew);
      Ps[h][v + 8 * half][l16]      = (__bf16)e0;   // keys 0..15
      Ps[h][v + 8 * half][16 + l16] = (__bf16)e1;   // keys 16..31
    }

    // first key half: l += P@1, out += P@V
    {
      BF16Frag pf;
      pf.h[0] = *(const bf16x8*)(&Ps[h][l16][16 * half]);
      pf.h[1] = *(const bf16x8*)(&Ps[h][l16][16 * half + 8]);
      accl = __builtin_amdgcn_wmma_f32_16x16x32_bf16(
          false, pf.v, false, ones.v, (short)0, accl, false, false);
#pragma unroll
      for (int dt = 0; dt < 8; ++dt) {
        BF16Frag vb;
        const __bf16* src = &Vts[dt * 16 + l16][16 * half];
        vb.h[0] = *(const bf16x8*)(src);
        vb.h[1] = *(const bf16x8*)(src + 8);
        acc[dt] = __builtin_amdgcn_wmma_f32_16x16x32_bf16(
            false, pf.v, false, vb.v, (short)0, acc[dt], false, false);
      }
    }
    // second key half (per-wave DS ops are in-order, safe to reuse Ps)
#pragma unroll
    for (int v = 0; v < 8; ++v) {
      Ps[h][v + 8 * half][l16]      = (__bf16)pA[v];  // keys 32..47
      Ps[h][v + 8 * half][16 + l16] = (__bf16)pB[v];  // keys 48..63
    }
    {
      BF16Frag pf;
      pf.h[0] = *(const bf16x8*)(&Ps[h][l16][16 * half]);
      pf.h[1] = *(const bf16x8*)(&Ps[h][l16][16 * half + 8]);
      accl = __builtin_amdgcn_wmma_f32_16x16x32_bf16(
          false, pf.v, false, ones.v, (short)0, accl, false, false);
#pragma unroll
      for (int dt = 0; dt < 8; ++dt) {
        BF16Frag vb;
        const __bf16* src = &Vts[dt * 16 + l16][32 + 16 * half];
        vb.h[0] = *(const bf16x8*)(src);
        vb.h[1] = *(const bf16x8*)(src + 8);
        acc[dt] = __builtin_amdgcn_wmma_f32_16x16x32_bf16(
            false, pf.v, false, vb.v, (short)0, acc[dt], false, false);
      }
    }
    __syncthreads();
  }

#pragma unroll
  for (int v = 0; v < 8; ++v) {
    float inv = 1.0f / accl[v];       // all 16 cols replicated -> lane-local ok
#pragma unroll
    for (int dt = 0; dt < 8; ++dt) {
      int col = h * 128 + dt * 16 + l16;
      O[(size_t)(row0 + v + 8 * half) * 2048 + col] = (__bf16)(acc[dt][v] * inv);
    }
  }
}

// ---------------------------------------------------------------------------
extern "C" void kernel_launch(void* const* d_in, const int* in_sizes, int n_in,
                              void* d_out, int out_size, void* d_ws, size_t ws_size,
                              hipStream_t stream) {
  (void)in_sizes; (void)n_in; (void)out_size; (void)ws_size;
  const float* x  = (const float*)d_in[0];
  const float* Wq = (const float*)d_in[1];
  const float* bq = (const float*)d_in[2];
  const float* Wk = (const float*)d_in[3];
  const float* bk = (const float*)d_in[4];
  const float* Wv = (const float*)d_in[5];
  const float* bv = (const float*)d_in[6];
  const float* Wo = (const float*)d_in[7];
  const float* bo = (const float*)d_in[8];
  float* out = (float*)d_out;

  const int M = 4096;           // B*S
  const int H = 2048, D = 128, S = 2048;
  const float kscale = 0.08838834764831845f;  // 1/sqrt(128)

  __bf16* q    = (__bf16*)d_ws;              // 16 MB
  __bf16* k    = q + (size_t)M * H;          //  1 MB
  __bf16* v    = k + (size_t)M * D;          //  1 MB
  __bf16* attn = v + (size_t)M * D;          // 16 MB

  dim3 blk(256);
  gemm_bf16_wmma<float, __bf16><<<dim3(H / 128, M / 256), blk, 0, stream>>>(x, Wq, bq, q, M, H, H, 1.0f);
  gemm_bf16_wmma<float, __bf16><<<dim3(D / 128, M / 256), blk, 0, stream>>>(x, Wk, bk, k, M, D, H, kscale);
  gemm_bf16_wmma<float, __bf16><<<dim3(D / 128, M / 256), blk, 0, stream>>>(x, Wv, bv, v, M, D, H, 1.0f);
  mqa_flash<<<dim3(M / 16), dim3(512), 0, stream>>>(q, k, v, attn, S);
  gemm_bf16_wmma<__bf16, float><<<dim3(H / 128, M / 256), blk, 0, stream>>>(attn, Wo, bo, out, M, H, H, 1.0f);
}
